// ArcFaceLoss_90915867722128
// MI455X (gfx1250) — matrix operand order repaired and checked by reading, not compiled, not run on hardware
//
#include <hip/hip_runtime.h>
#include <hip/hip_bf16.h>
#include <math.h>

typedef __attribute__((ext_vector_type(16))) _Float16 v16h;
typedef __attribute__((ext_vector_type(8)))  float    v8f;

#define S_SCALE     30.0f
#define MARGIN      0.5f
#define NUM_CLASSES 50000
#define EMB         512
#define BATCH       2048
#define ARC_EPS     1e-7f

#define ROW_TILES   (BATCH / 16)          // 128
#define COL_TILES   (NUM_CLASSES / 16)    // 3125
#define KSTEPS      (EMB / 32)            // 16
#define FRAG_HALVES (KSTEPS * 32 * 16)    // 8192 halves per 16xEMB tile
#define NCHUNKS     64
#define TILES_PER_CHUNK ((COL_TILES + NCHUNKS - 1) / NCHUNKS)  // 49
#define WAVES       8
#define MTILES      4                     // 64 rows per block
#define SUPER_ROWS  (ROW_TILES / MTILES)  // 32

// ---------------------------------------------------------------------------
// Prep A: normalize embedding rows, emit f16 WMMA A-fragments (16x32 layout).
// A layout (ISA 7.12.2, 16-bit A 16x32): lane<16: M=lane, half j -> K = j<8? j : j+8
//                                        lane>=16: M=lane-16,  half j -> K = j<8? j+8 : j+16
// ---------------------------------------------------------------------------
__global__ __launch_bounds__(256)
void prep_a(const float* __restrict__ emb, _Float16* __restrict__ aFrag) {
    __shared__ float x[16 * EMB];     // 32 KB
    __shared__ float part[256];
    __shared__ float invn[16];
    const int tile = blockIdx.x;
    const int t = threadIdx.x;

    const float* src = emb + (size_t)tile * 16 * EMB;
    for (int i = t; i < 16 * EMB; i += 256) x[i] = src[i];
    __syncthreads();

    // 16 threads per row compute partial sum of squares
    {
        const int r = t >> 4, sub = t & 15;
        const float* row = x + r * EMB + sub * 32;
        float acc = 0.f;
        #pragma unroll
        for (int j = 0; j < 32; ++j) acc += row[j] * row[j];
        part[t] = acc;
    }
    __syncthreads();
    if (t < 16) {
        float s = 0.f;
        for (int j = 0; j < 16; ++j) s += part[t * 16 + j];
        invn[t] = 1.0f / fmaxf(sqrtf(s), 1e-12f);
    }
    __syncthreads();

    _Float16* dst = aFrag + (size_t)tile * FRAG_HALVES;
    for (int F = t * 32; F < t * 32 + 32; ++F) {
        const int j    = F & 15;
        const int lane = (F >> 4) & 31;
        const int ks   = F >> 9;
        int rr, kk;
        if (lane < 16) { rr = lane;      kk = (j < 8) ? j       : (j + 8);  }
        else           { rr = lane - 16; kk = (j < 8) ? (j + 8) : (j + 16); }
        const int K = ks * 32 + kk;
        dst[F] = (_Float16)(x[rr * EMB + K] * invn[rr]);
    }
}

// ---------------------------------------------------------------------------
// Prep B: normalize class rows, emit f16 WMMA B-fragments (32x16 layout).
// B layout: lane -> N = lane&15 ; half j -> K = (j<8 ? j : j+8) + 8*(lane>=16)
// ---------------------------------------------------------------------------
__global__ __launch_bounds__(256)
void prep_b(const float* __restrict__ wgt, _Float16* __restrict__ bFrag) {
    __shared__ float x[16 * EMB];     // 16 classes x 512
    __shared__ float part[256];
    __shared__ float invn[16];
    const int tile = blockIdx.x;
    const int t = threadIdx.x;

    const float* src = wgt + (size_t)tile * 16 * EMB;
    for (int i = t; i < 16 * EMB; i += 256) x[i] = src[i];
    __syncthreads();

    {
        const int r = t >> 4, sub = t & 15;
        const float* row = x + r * EMB + sub * 32;
        float acc = 0.f;
        #pragma unroll
        for (int j = 0; j < 32; ++j) acc += row[j] * row[j];
        part[t] = acc;
    }
    __syncthreads();
    if (t < 16) {
        float s = 0.f;
        for (int j = 0; j < 16; ++j) s += part[t * 16 + j];
        invn[t] = 1.0f / fmaxf(sqrtf(s), 1e-12f);
    }
    __syncthreads();

    _Float16* dst = bFrag + (size_t)tile * FRAG_HALVES;
    for (int F = t * 32; F < t * 32 + 32; ++F) {
        const int j    = F & 15;
        const int lane = (F >> 4) & 31;
        const int ks   = F >> 9;
        const int N  = lane & 15;
        const int kk = ((j < 8) ? j : (j + 8)) + ((lane >> 4) << 3);
        const int K  = ks * 32 + kk;
        dst[F] = (_Float16)(x[N * EMB + K] * invn[N]);
    }
}

// ---------------------------------------------------------------------------
// Fused GEMM + online softmax. Block: 64 rows (4 M-tiles) x one class chunk.
// ---------------------------------------------------------------------------
__global__ __launch_bounds__(256)
void arc_gemm(const _Float16* __restrict__ aFrag,
              const _Float16* __restrict__ bFrag,
              const int* __restrict__ labels,
              float* __restrict__ partM, float* __restrict__ partS,
              float* __restrict__ logitLab) {
    __shared__ _Float16 shA[MTILES * FRAG_HALVES];   // 64 KB
    __shared__ float shM[WAVES][MTILES * 16];        // 2 KB
    __shared__ float shS[WAVES][MTILES * 16];        // 2 KB
    __shared__ int   shLab[MTILES * 16];

    const int superTile = blockIdx.x;                // 0..31
    const int chunk     = blockIdx.y;                // 0..63
    const int t    = threadIdx.x;
    const int wv   = t >> 5;
    const int lane = t & 31;
    const int rowBase = superTile * (MTILES * 16);

    // Stage A fragments for 4 row tiles into LDS (64 KB, reused by all waves/ksteps)
    {
        const uint4* src = (const uint4*)(aFrag + (size_t)superTile * MTILES * FRAG_HALVES);
        uint4* dst = (uint4*)shA;
        const int n16 = (MTILES * FRAG_HALVES * 2) / 16;   // 4096
        for (int i = t; i < n16; i += 256) dst[i] = src[i];
    }
    if (t < MTILES * 16) shLab[t] = labels[rowBase + t];
    __syncthreads();

    const int t0 = chunk * TILES_PER_CHUNK;
    const int t1 = min(t0 + TILES_PER_CHUNK, COL_TILES);

    float m[MTILES][8], s[MTILES][8];
    #pragma unroll
    for (int mt = 0; mt < MTILES; ++mt)
        #pragma unroll
        for (int v = 0; v < 8; ++v) { m[mt][v] = -1e30f; s[mt][v] = 0.f; }

    const int myRow0 = (lane >> 4) * 8;   // C layout: lanes 16-31 hold M = v+8
    const int myCol  = lane & 15;

    for (int ct = t0 + wv; ct < t1; ct += WAVES) {     // wave-uniform: EXEC all-1s for WMMA
        // Defeat LICM: an opaque (asm-modified) zero offset makes the LDS
        // addresses loop-variant, so A fragments are re-read from LDS via
        // ds_load each iteration instead of being hoisted (which previously
        // spilled 512 VGPRs of A fragments to scratch).
        unsigned opaque = 0;
        asm volatile("" : "+s"(opaque));
        const _Float16* ap = shA + lane * 16 + opaque;

        v8f acc0 = {}, acc1 = {}, acc2 = {}, acc3 = {};
        const _Float16* bp = bFrag + (size_t)ct * FRAG_HALVES + lane * 16;

        const int nt = ct + WAVES;                     // prefetch next B tile
        if (nt < t1) __builtin_prefetch(bFrag + (size_t)nt * FRAG_HALVES + lane * 16, 0, 1);

        #pragma unroll
        for (int ks = 0; ks < KSTEPS; ++ks) {
            const v16h b  = *(const v16h*)(bp + ks * 512);
            const v16h a0 = *(const v16h*)(ap + 0 * FRAG_HALVES + ks * 512);
            acc0 = __builtin_amdgcn_wmma_f32_16x16x32_f16(false, a0, false, b, (short)0, acc0, false, false);
            const v16h a1 = *(const v16h*)(ap + 1 * FRAG_HALVES + ks * 512);
            acc1 = __builtin_amdgcn_wmma_f32_16x16x32_f16(false, a1, false, b, (short)0, acc1, false, false);
            const v16h a2 = *(const v16h*)(ap + 2 * FRAG_HALVES + ks * 512);
            acc2 = __builtin_amdgcn_wmma_f32_16x16x32_f16(false, a2, false, b, (short)0, acc2, false, false);
            const v16h a3 = *(const v16h*)(ap + 3 * FRAG_HALVES + ks * 512);
            acc3 = __builtin_amdgcn_wmma_f32_16x16x32_f16(false, a3, false, b, (short)0, acc3, false, false);
        }

        const int colG = ct * 16 + myCol;
        #pragma unroll
        for (int mt = 0; mt < MTILES; ++mt) {
            const v8f acc = (mt == 0) ? acc0 : (mt == 1) ? acc1 : (mt == 2) ? acc2 : acc3;
            #pragma unroll
            for (int v = 0; v < 8; ++v) {
                const int rloc = mt * 16 + myRow0 + v;
                const float c = acc[v];
                float logit;
                if (shLab[rloc] == colG) {
                    const float cc = fminf(fmaxf(c, -1.0f + ARC_EPS), 1.0f - ARC_EPS);
                    logit = S_SCALE * cosf(acosf(cc) + MARGIN);
                    logitLab[rowBase + rloc] = logit;   // exactly one writer per row
                } else {
                    logit = S_SCALE * c;
                }
                if (logit > m[mt][v]) {
                    s[mt][v] = s[mt][v] * expf(m[mt][v] - logit) + 1.0f;
                    m[mt][v] = logit;
                } else {
                    s[mt][v] += expf(logit - m[mt][v]);
                }
            }
        }
    }

    // merge (m,s) across the 16 lanes that hold the same rows
    #pragma unroll
    for (int mt = 0; mt < MTILES; ++mt)
        #pragma unroll
        for (int v = 0; v < 8; ++v)
            for (int off = 1; off < 16; off <<= 1) {
                const float m2 = __shfl_xor(m[mt][v], off, 32);
                const float s2 = __shfl_xor(s[mt][v], off, 32);
                const float M  = fmaxf(m[mt][v], m2);
                s[mt][v] = s[mt][v] * expf(m[mt][v] - M) + s2 * expf(m2 - M);
                m[mt][v] = M;
            }

    if ((lane & 15) == 0) {
        #pragma unroll
        for (int mt = 0; mt < MTILES; ++mt)
            #pragma unroll
            for (int v = 0; v < 8; ++v) {
                shM[wv][mt * 16 + myRow0 + v] = m[mt][v];
                shS[wv][mt * 16 + myRow0 + v] = s[mt][v];
            }
    }
    __syncthreads();

    if (t < MTILES * 16) {      // one thread per row: merge 8 waves, write chunk partial
        float M = -1e30f, Sm = 0.f;
        for (int w = 0; w < WAVES; ++w) {
            const float m2 = shM[w][t], s2 = shS[w][t];
            const float MM = fmaxf(M, m2);
            Sm = Sm * expf(M - MM) + s2 * expf(m2 - MM);
            M = MM;
        }
        partM[(size_t)(rowBase + t) * NCHUNKS + chunk] = M;
        partS[(size_t)(rowBase + t) * NCHUNKS + chunk] = Sm;
    }
}

// ---------------------------------------------------------------------------
// Finalize: per-row logsumexp over chunks, NLL, mean.
// ---------------------------------------------------------------------------
__global__ __launch_bounds__(256)
void arc_finalize(const float* __restrict__ partM, const float* __restrict__ partS,
                  const float* __restrict__ logitLab, float* __restrict__ out) {
    __shared__ float red[256];
    const int t = threadIdx.x;
    float acc = 0.f;
    for (int r = t; r < BATCH; r += 256) {
        float M = -1e30f, Sm = 0.f;
        for (int c = 0; c < NCHUNKS; ++c) {
            const float m2 = partM[(size_t)r * NCHUNKS + c];
            const float s2 = partS[(size_t)r * NCHUNKS + c];
            const float MM = fmaxf(M, m2);
            Sm = Sm * expf(M - MM) + s2 * expf(m2 - MM);
            M = MM;
        }
        acc += (M + logf(Sm)) - logitLab[r];
    }
    red[t] = acc;
    __syncthreads();
    for (int off = 128; off > 0; off >>= 1) {
        if (t < off) red[t] += red[t + off];
        __syncthreads();
    }
    if (t == 0) out[0] = red[0] / (float)BATCH;
}

// ---------------------------------------------------------------------------
extern "C" void kernel_launch(void* const* d_in, const int* in_sizes, int n_in,
                              void* d_out, int out_size, void* d_ws, size_t ws_size,
                              hipStream_t stream) {
    const float* emb    = (const float*)d_in[0];
    const int*   labels = (const int*)d_in[1];
    const float* weight = (const float*)d_in[2];
    float* out = (float*)d_out;
    char*  ws  = (char*)d_ws;

    const size_t szB = (size_t)COL_TILES * FRAG_HALVES * sizeof(_Float16); // 51.2 MB
    const size_t szA = (size_t)ROW_TILES * FRAG_HALVES * sizeof(_Float16); // 2 MB
    const size_t szP = (size_t)BATCH * NCHUNKS * sizeof(float);            // 512 KB

    _Float16* bFrag    = (_Float16*)(ws);
    _Float16* aFrag    = (_Float16*)(ws + szB);
    float*    partM    = (float*)(ws + szB + szA);
    float*    partS    = (float*)(ws + szB + szA + szP);
    float*    logitLab = (float*)(ws + szB + szA + 2 * szP);

    prep_a<<<ROW_TILES, 256, 0, stream>>>(emb, aFrag);
    prep_b<<<COL_TILES, 256, 0, stream>>>(weight, bFrag);
    arc_gemm<<<dim3(SUPER_ROWS, NCHUNKS), 256, 0, stream>>>(aFrag, bFrag, labels,
                                                            partM, partS, logitLab);
    arc_finalize<<<1, 256, 0, stream>>>(partM, partS, logitLab, out);
}
